// VisionDeltaNet_34342558498951
// MI455X (gfx1250) — matrix operand order repaired and checked by reading, not compiled, not run on hardware
//
#include <hip/hip_runtime.h>

// ---------------------------------------------------------------------------
// VisionDeltaNet forward for MI455X (gfx1250): wave32, bf16 WMMA, TDM staging.
// ---------------------------------------------------------------------------

#define DEV __device__ __forceinline__

typedef __bf16 bf16_t;
typedef bf16_t v8bf  __attribute__((ext_vector_type(8)));
typedef bf16_t v16bf __attribute__((ext_vector_type(16)));
typedef float  v8f   __attribute__((ext_vector_type(8)));
typedef unsigned int u32x4 __attribute__((ext_vector_type(4)));
typedef int i32x4 __attribute__((ext_vector_type(4)));
typedef int i32x8 __attribute__((ext_vector_type(8)));

static constexpr int HD = 384;     // hidden
static constexpr int ED = 768;     // attn dim
static constexpr int LD = 1025;    // tokens
static constexpr int BD = 8;       // batch
static constexpr long BL = (long)BD * LD;  // 8200 rows

#if defined(__has_builtin)
#if __has_builtin(__builtin_amdgcn_tensor_load_to_lds) && \
    __has_builtin(__builtin_amdgcn_s_wait_tensorcnt)
#define HAVE_TDM 1
#endif
#endif

DEV unsigned int f2bf(float f) {
  unsigned int u = __float_as_uint(f);
  u += 0x7FFFu + ((u >> 16) & 1u);          // round-to-nearest-even
  return u >> 16;
}
DEV unsigned int pack2bf(float lo, float hi) {
  return f2bf(lo) | (f2bf(hi) << 16);
}
DEV float gelu_f(float x) { return 0.5f * x * (1.f + erff(x * 0.70710678118654752f)); }
DEV float sigm_f(float x) { return 1.f / (1.f + expf(-x)); }

#if HAVE_TDM
// ---------------------------------------------------------------------------
// Tensor Data Mover: DMA a fully in-bounds (rows x 32) bf16 tile (row stride
// `rowStride` elements) from global memory into LDS at byte offset ldsOff.
// D# per CDNA5 ISA: group0 = {count=1, lds_addr,
// global_addr, type=2}; group1 = {data_size=2B, tensor dims, tile dims,
// dim0 stride}. Groups 2/3 unused (2D tensor).
// ---------------------------------------------------------------------------
DEV void tdm_load_tile(const unsigned short* gp, unsigned ldsOff, int rows,
                       long rowStride)
{
  const unsigned long long ga = (unsigned long long)(const void*)gp;
  u32x4 g0;
  g0[0] = 1u;                                            // count=1 (valid D#)
  g0[1] = ldsOff;                                        // LDS byte address
  g0[2] = (unsigned)ga;                                  // global_addr[31:0]
  g0[3] = (unsigned)((ga >> 32) & 0x01FFFFFFu) | (2u << 30); // addr[56:32]|type=2
  const unsigned d0 = 32u;                 // tensor_dim0 (k extent)
  const unsigned d1 = (unsigned)rows;      // tensor_dim1 (row extent)
  const unsigned long long s0 = (unsigned long long)rowStride;
  i32x8 g1;
  g1[0] = 1 << 16;                                  // wg_mask=0, data_size=2B
  g1[1] = (int)(d0 << 16);                          // tensor_dim0[15:0]
  g1[2] = (int)((d0 >> 16) | (d1 << 16));           // dim0 hi | dim1 lo
  g1[3] = (int)((d1 >> 16) | (32u << 16));          // dim1 hi | tile_dim0=32
  g1[4] = (int)d1;                                  // tile_dim1=rows, tile_dim2=0
  g1[5] = (int)(unsigned)(s0 & 0xFFFFFFFFu);        // dim0_stride lo
  g1[6] = (int)(unsigned)((s0 >> 32) & 0xFFFFu);    // dim0_stride hi | dim1_stride=0
  g1[7] = 0;
  i32x4 zg; zg[0] = 0; zg[1] = 0; zg[2] = 0; zg[3] = 0;
#if __clang_major__ >= 23
  i32x8 zg8;
#pragma unroll
  for (int i = 0; i < 8; ++i) zg8[i] = 0;
  __builtin_amdgcn_tensor_load_to_lds(g0, g1, zg, zg, zg8, 0);
#else
  __builtin_amdgcn_tensor_load_to_lds(g0, g1, zg, zg, 0);
#endif
}
#endif // HAVE_TDM

// ---------------------------------------------------------------------------
// Fallback staging of a (rows x 32) bf16 tile into LDS (row-major, K contig).
// ks==1 -> uint4 loads along K; rs==1 -> uint4 loads along M (transposed store).
// ---------------------------------------------------------------------------
union U8 { uint4 v; unsigned short u[8]; };

DEV void stage_tile_bf(const unsigned short* __restrict__ P, long rs, long ks,
                       int g0, int GM, int k0, int K,
                       unsigned short* lds, int rows, int tid)
{
  const int slots = rows * 4;                        // 8 elements per slot
  if (ks == 1) {
    const bool fullK = (k0 + 32 <= K);
    for (int s = tid; s < slots; s += 128) {
      const int mm = s >> 2;
      const int kk = (s & 3) << 3;
      const int gm = g0 + mm, gk = k0 + kk;
      U8 w; w.v.x = 0u; w.v.y = 0u; w.v.z = 0u; w.v.w = 0u;
      if (gm < GM) {
        const unsigned short* p = P + (long)gm * rs + gk;
        if (fullK) w.v = *(const uint4*)p;
        else {
#pragma unroll
          for (int e = 0; e < 8; ++e) w.u[e] = (gk + e < K) ? p[e] : (unsigned short)0;
        }
      }
      *(uint4*)&lds[mm * 32 + kk] = w.v;
    }
  } else if (rs == 1) {
    for (int s = tid; s < slots; s += 128) {
      const int kk = s & 31;
      const int mm = (s >> 5) << 3;
      const int gk = k0 + kk;
      U8 w; w.v.x = 0u; w.v.y = 0u; w.v.z = 0u; w.v.w = 0u;
      if (gk < K) {
        const int gm = g0 + mm;
        const unsigned short* p = P + (long)gk * ks + gm;
        if (gm + 7 < GM) w.v = *(const uint4*)p;
        else {
#pragma unroll
          for (int e = 0; e < 8; ++e) w.u[e] = (gm + e < GM) ? p[e] : (unsigned short)0;
        }
      }
#pragma unroll
      for (int e = 0; e < 8; ++e) lds[(mm + e) * 32 + kk] = w.u[e];
    }
  } else {
    for (int i = tid; i < rows * 32; i += 128) {
      const int mm = i >> 5, kk = i & 31;
      const int gm = g0 + mm, gk = k0 + kk;
      lds[i] = (gm < GM && gk < K) ? P[(long)gm * rs + (long)gk * ks]
                                   : (unsigned short)0;
    }
  }
}

// ---------------------------------------------------------------------------
// Generic strided batched GEMM:  C[m,n] = epi( sum_k A[m,k] * W[n,k] + bias[n] )
// A/W are bf16 with element strides (rs, ks) (+ z*sb). C f32 or bf16 (outBf).
// epi: act 0=none, 1=gelu, 2=sigmoid*0.9+0.1 ; then += add[m*ldAdd+n] if add.
// 128x64 C tile / workgroup, 4 waves x (32x64), K-step 32, bf16 WMMA, f32 acc.
// Full interior tiles are staged by the Tensor Data Mover (async DMA to LDS).
// ---------------------------------------------------------------------------
__global__ __launch_bounds__(128) void gemm_bf16_wmma(
    const unsigned short* __restrict__ A, long rsA, long ksA, long sbA,
    const unsigned short* __restrict__ W, long rsB, long ksB, long sbB,
    const float* __restrict__ bias,
    const float* __restrict__ add, long ldAdd, long sbAdd,
    float* __restrict__ C, long ldC, long sbC,
    int M, int N, int K, int act, int outBf)
{
  __shared__ __align__(32) unsigned short ldsA[128 * 32];
  __shared__ __align__(32) unsigned short ldsB[64 * 32];

  const int tid  = threadIdx.x;
  const int lane = tid & 31;
  const int wv   = tid >> 5;          // wave 0..3 -> C rows 32*wv..32*wv+31
  const int h16  = lane >> 4;         // lane half
  const int lr   = lane & 15;
  const int m0 = blockIdx.y * 128;
  const int n0 = blockIdx.x * 64;
  const int bz = blockIdx.z;

  A += (long)bz * sbA;
  W += (long)bz * sbB;
  if (add) add += (long)bz * sbAdd;

#if HAVE_TDM
  const unsigned ldsOffA = (unsigned)(unsigned long long)(const void*)ldsA;
  const unsigned ldsOffB = (unsigned)(unsigned long long)(const void*)ldsB;
#endif

  v8f acc[2][4];
#pragma unroll
  for (int s = 0; s < 2; ++s)
#pragma unroll
    for (int t = 0; t < 4; ++t)
#pragma unroll
      for (int r = 0; r < 8; ++r) acc[s][t][r] = 0.f;

  for (int k0 = 0; k0 < K; k0 += 32) {
    const bool fullK = (k0 + 32 <= K);
    bool tdmA = false, tdmB = false;
#if HAVE_TDM
    tdmA = (ksA == 1) && fullK && (m0 + 128 <= M);
    tdmB = (ksB == 1) && fullK && (n0 + 64 <= N);
    if ((tdmA || tdmB) && wv == 0) {
      if (tdmA) tdm_load_tile(A + (long)m0 * rsA + k0, ldsOffA, 128, rsA);
      if (tdmB) tdm_load_tile(W + (long)n0 * rsB + k0, ldsOffB, 64, rsB);
      __builtin_amdgcn_s_wait_tensorcnt(0);
    }
#endif
    if (!tdmA) stage_tile_bf(A, rsA, ksA, m0, M, k0, K, ldsA, 128, tid);
    if (!tdmB) stage_tile_bf(W, rsB, ksB, n0, N, k0, K, ldsB, 64, tid);
    __syncthreads();

    // A fragments (16x32 each): lane holds row, K runs {8h..8h+7},{16+8h..}
    v16bf af[2];
#pragma unroll
    for (int s = 0; s < 2; ++s) {
      const unsigned short* ap = &ldsA[(32 * wv + 16 * s + lr) * 32 + 8 * h16];
      const v8bf a0 = *(const v8bf*)ap;
      const v8bf a1 = *(const v8bf*)(ap + 16);
#pragma unroll
      for (int i = 0; i < 8; ++i) { af[s][i] = a0[i]; af[s][i + 8] = a1[i]; }
    }

#pragma unroll
    for (int t = 0; t < 4; ++t) {
      // B fragment (32x16): lane holds column n=lr, K runs {8h..8h+7},{16+8h..}
      const unsigned short* bp = &ldsB[(16 * t + lr) * 32 + 8 * h16];
      const v8bf b0 = *(const v8bf*)bp;
      const v8bf b1 = *(const v8bf*)(bp + 16);
      v16bf bfr;
#pragma unroll
      for (int i = 0; i < 8; ++i) { bfr[i] = b0[i]; bfr[i + 8] = b1[i]; }
      acc[0][t] = __builtin_amdgcn_wmma_f32_16x16x32_bf16(
          false, af[0], false, bfr, (short)0, acc[0][t], false, false);
      acc[1][t] = __builtin_amdgcn_wmma_f32_16x16x32_bf16(
          false, af[1], false, bfr, (short)0, acc[1][t], false, false);
    }
    __syncthreads();
  }

  // epilogue: C/D layout -> reg r, lane-half h16: m = 32wv+16s+8h16+r, n = 16t+lr
  float* Cf = C + (long)bz * sbC;
  unsigned short* Cb = (unsigned short*)C + (long)bz * sbC;
#pragma unroll
  for (int t = 0; t < 4; ++t) {
    const int n = n0 + 16 * t + lr;
    if (n >= N) continue;
    const float bi = bias ? bias[n] : 0.f;
#pragma unroll
    for (int s = 0; s < 2; ++s) {
#pragma unroll
      for (int r = 0; r < 8; ++r) {
        const int m = m0 + 32 * wv + 16 * s + 8 * h16 + r;
        if (m >= M) continue;
        float val = acc[s][t][r] + bi;
        if (act == 1)      val = gelu_f(val);
        else if (act == 2) val = sigm_f(val) * 0.9f + 0.1f;
        if (add) val += add[(long)m * ldAdd + n];
        if (outBf) Cb[(long)m * ldC + n] = (unsigned short)f2bf(val);
        else       Cf[(long)m * ldC + n] = val;
      }
    }
  }
}

// f32 -> bf16 bulk conversion (n multiple of 4)
__global__ __launch_bounds__(256) void cvt_bf16_kernel(
    const float* __restrict__ src, unsigned short* __restrict__ dst, long n)
{
  const long i = ((long)blockIdx.x * 256 + threadIdx.x) * 4;
  if (i >= n) return;
  const float4 v = *(const float4*)(src + i);
  uint2 pk;
  pk.x = pack2bf(v.x, v.y);
  pk.y = pack2bf(v.z, v.w);
  *(uint2*)&dst[i] = pk;
}

// ---------------------------------------------------------------------------
// LayerNorm, one wave per row; reads f32, writes bf16. Optional pre-scale.
// ---------------------------------------------------------------------------
__global__ __launch_bounds__(256) void ln_kernel(
    const float* __restrict__ src, long srcRow,
    unsigned short* __restrict__ dst, long dstRow,
    const float* __restrict__ g, const float* __restrict__ b,
    const float* __restrict__ scalePtr, int rows, int dim)
{
  const int row = blockIdx.x * 8 + (threadIdx.x >> 5);
  if (row >= rows) return;
  const int lane = threadIdx.x & 31;
  const float* x = src + (long)row * srcRow;
  unsigned short* y = dst + (long)row * dstRow;
  const float s = scalePtr ? scalePtr[0] : 1.f;

  float sum = 0.f, sq = 0.f;
  for (int i = lane; i < dim; i += 32) {
    float v = x[i] * s;
    sum += v; sq += v * v;
  }
#pragma unroll
  for (int o = 16; o; o >>= 1) {
    sum += __shfl_xor(sum, o, 32);
    sq  += __shfl_xor(sq,  o, 32);
  }
  const float m = sum / dim;
  const float var = sq / dim - m * m;
  const float r = rsqrtf(var + 1e-5f);
  for (int i = lane; i < dim; i += 32) {
    float v = x[i] * s;
    y[i] = (unsigned short)f2bf((v - m) * r * g[i] + b[i]);
  }
}

// ---------------------------------------------------------------------------
// QKV activation + coupled l2norm, in-place f32 on (rows x 3E) buffer.
// q=silu(q); k=silu(k); v=gelu(v); q=l2n(q)+0.1k; k=l2n(k)+0.1q_new
// ---------------------------------------------------------------------------
__global__ __launch_bounds__(256) void qkvact_kernel(float* __restrict__ qkv, int rows)
{
  const int row = blockIdx.x * 8 + (threadIdx.x >> 5);
  if (row >= rows) return;
  const int lane = threadIdx.x & 31;
  float* base = qkv + (long)row * (3 * ED);

  float q[24], k[24], v[24];
  float sq = 0.f, sk = 0.f;
#pragma unroll
  for (int j = 0; j < 24; ++j) {
    const int i = lane + 32 * j;
    float qv = base[i], kv = base[ED + i], vv = base[2 * ED + i];
    qv = qv * sigm_f(qv);
    kv = kv * sigm_f(kv);
    vv = gelu_f(vv);
    q[j] = qv; k[j] = kv; v[j] = vv;
    sq += qv * qv; sk += kv * kv;
  }
#pragma unroll
  for (int o = 16; o; o >>= 1) {
    sq += __shfl_xor(sq, o, 32);
    sk += __shfl_xor(sk, o, 32);
  }
  const float iq = 1.f / fmaxf(sqrtf(sq), 1e-12f);
  const float ik = 1.f / fmaxf(sqrtf(sk), 1e-12f);
#pragma unroll
  for (int j = 0; j < 24; ++j) {
    const int i = lane + 32 * j;
    const float qn = q[j] * iq + 0.1f * k[j];
    const float kn = k[j] * ik + 0.1f * qn;
    base[i] = qn; base[ED + i] = kn; base[2 * ED + i] = v[j];
  }
}

// Depthwise 3-tap conv along tokens + fused beta-gating of v; writes bf16.
__global__ __launch_bounds__(256) void dwconv_kernel(
    const float* __restrict__ x, unsigned short* __restrict__ y,
    const float* __restrict__ cw, const float* __restrict__ beta, int total)
{
  const int j = blockIdx.x * 256 + threadIdx.x;
  if (j >= total) return;
  const int E3 = 3 * ED;
  const int ch3 = j % E3;
  const long bl = j / E3;
  const int l = (int)(bl % LD);
  const float* w = cw + (ch3 % ED) * 3;
  float acc = w[1] * x[j];
  if (l > 0)      acc += w[0] * x[j - E3];
  if (l + 1 < LD) acc += w[2] * x[j + E3];
  if (ch3 >= 2 * ED) acc *= beta[bl * ED + (ch3 - 2 * ED)];   // v_new = beta*v
  y[j] = (unsigned short)f2bf(acc);
}

// conv1: (8,3,128,128) -> gelu(conv 3x3 s2 p1) -> (8,192,64,64) f32
__global__ __launch_bounds__(256) void conv1_kernel(
    const float* __restrict__ x, const float* __restrict__ wgt,
    const float* __restrict__ bias, float* __restrict__ out)
{
  const int idx = blockIdx.x * 256 + threadIdx.x;
  const int OC = 192, OH = 64, OW = 64, IH = 128, IW = 128;
  if (idx >= BD * OC * OH * OW) return;
  const int ox = idx % OW, oy = (idx / OW) % OH;
  const int oc = (idx / (OW * OH)) % OC, b = idx / (OW * OH * OC);
  float acc = bias[oc];
#pragma unroll
  for (int ic = 0; ic < 3; ++ic)
#pragma unroll
    for (int ky = 0; ky < 3; ++ky) {
      const int iy = 2 * oy + ky - 1;
      if (iy < 0 || iy >= IH) continue;
#pragma unroll
      for (int kx = 0; kx < 3; ++kx) {
        const int ix = 2 * ox + kx - 1;
        if (ix < 0 || ix >= IW) continue;
        acc += x[((b * 3 + ic) * IH + iy) * IW + ix] *
               wgt[((oc * 3 + ic) * 3 + ky) * 3 + kx];
      }
    }
  out[idx] = gelu_f(acc);
}

// im2col for conv2: (8,192,64,64) f32 -> (8,1024,1728) bf16
__global__ __launch_bounds__(256) void im2col_kernel(
    const float* __restrict__ h1, unsigned short* __restrict__ col)
{
  const int idx = blockIdx.x * 256 + threadIdx.x;
  const int total = BD * 1024 * 1728;
  if (idx >= total) return;
  const int k = idx % 1728;
  const int t = (idx / 1728) % 1024;
  const int b = idx / (1728 * 1024);
  const int kx = k % 3, ky = (k / 3) % 3, ic = k / 9;
  const int ox = t % 32, oy = t / 32;
  const int iy = 2 * oy + ky - 1, ix = 2 * ox + kx - 1;
  float v = 0.f;
  if (iy >= 0 && iy < 64 && ix >= 0 && ix < 64)
    v = h1[((b * 192 + ic) * 64 + iy) * 64 + ix];
  col[idx] = (unsigned short)f2bf(v);
}

// cls token + positional embedding
__global__ __launch_bounds__(256) void embed_kernel(
    float* __restrict__ h, const float* __restrict__ cls, const float* __restrict__ pos)
{
  const int idx = blockIdx.x * 256 + threadIdx.x;
  if (idx >= BD * LD * HD) return;
  const int c = idx % HD;
  const int t = (idx / HD) % LD;
  if (t == 0) h[idx] = cls[c] + pos[c];
  else        h[idx] += pos[t * HD + c];
}

// ---------------------------------------------------------------------------
static void launch_gemm(hipStream_t s,
    const unsigned short* A, long rsA, long ksA, long sbA,
    const unsigned short* W, long rsB, long ksB, long sbB,
    const float* bias, const float* add, long ldAdd, long sbAdd,
    float* C, long ldC, long sbC, int M, int N, int K, int batch,
    int act, int outBf)
{
  dim3 grid((N + 63) / 64, (M + 127) / 128, batch);
  gemm_bf16_wmma<<<grid, 128, 0, s>>>(A, rsA, ksA, sbA, W, rsB, ksB, sbB,
                                      bias, add, ldAdd, sbAdd, C, ldC, sbC,
                                      M, N, K, act, outBf);
}

extern "C" void kernel_launch(void* const* d_in, const int* in_sizes, int n_in,
                              void* d_out, int out_size, void* d_ws, size_t ws_size,
                              hipStream_t stream) {
  const float* x        = (const float*)d_in[0];
  const float* pe1_w    = (const float*)d_in[1];
  const float* pe1_b    = (const float*)d_in[2];
  const float* pe2_w    = (const float*)d_in[3];
  const float* pe2_b    = (const float*)d_in[4];
  const float* cls_tok  = (const float*)d_in[5];
  const float* pos_emb  = (const float*)d_in[6];
  const float* ln1_g    = (const float*)d_in[7];
  const float* ln1_b    = (const float*)d_in[8];
  const float* ln2_g    = (const float*)d_in[9];
  const float* ln2_b    = (const float*)d_in[10];
  const float* win_w    = (const float*)d_in[11];
  const float* win_b    = (const float*)d_in[12];
  const float* wbeta_w  = (const float*)d_in[13];
  const float* wbeta_b  = (const float*)d_in[14];
  const float* wout_w   = (const float*)d_in[15];
  const float* wout_b   = (const float*)d_in[16];
  const float* mlp1_w   = (const float*)d_in[17];
  const float* mlp1_b   = (const float*)d_in[18];
  const float* mlp2_w   = (const float*)d_in[19];
  const float* mlp2_b   = (const float*)d_in[20];
  const float* conv_w   = (const float*)d_in[21];
  const float* attn_sc  = (const float*)d_in[22];
  const float* lnf_g    = (const float*)d_in[23];
  const float* lnf_b    = (const float*)d_in[24];
  const float* head_w   = (const float*)d_in[25];
  const float* head_b   = (const float*)d_in[26];
  float* out = (float*)d_out;

  // ---------------- workspace layout ----------------
  float* ws = (float*)d_ws;
  float* h     = ws;                     // B,L,H residual stream (f32)
  float* qkvA  = h     + 3148800L;       // qkv pre-dwconv f32 (alias: conv1 out)
  float* betaB = qkvA  + 18892800L;      // beta f32
  float* attnO = betaB + 6297600L;       // attention out f32
  float* h1    = qkvA;                   // conv1 output alias (6.29M f32)

  unsigned short* bfp = (unsigned short*)(attnO + 6297600L);
  size_t e = 0;
  unsigned short* xnBf    = bfp + e; e += BL * HD;             // LN output
  unsigned short* qkvBf   = bfp + e; e += BL * 3 * ED;         // qkv post-dwconv
  unsigned short* stateBf = bfp + e; e += (long)BD * ED * ED;  // state
  unsigned short* attnBf  = bfp + e; e += BL * ED;             // LN2 output
  unsigned short* midBf   = bfp + e; e += 14155776L;           // mlp mid / im2col
  unsigned short* clsnBf  = bfp + e; e += (long)BD * HD;
  unsigned short* winBf   = bfp + e; e += 12L * 3 * ED * HD;
  unsigned short* wbetaBf = bfp + e; e += 12L * ED * HD;
  unsigned short* woutBf  = bfp + e; e += 12L * HD * ED;
  unsigned short* m1Bf    = bfp + e; e += 12L * 4 * HD * HD;
  unsigned short* m2Bf    = bfp + e; e += 12L * 4 * HD * HD;
  unsigned short* pe2Bf   = bfp + e; e += 384L * 1728;
  unsigned short* headBf  = bfp + e; e += 1000L * 384;
  unsigned short* colBf   = midBf;

  const long E3 = 3 * ED;

  auto cvt = [&](const float* s, unsigned short* d, long n) {
    cvt_bf16_kernel<<<(int)((n / 4 + 255) / 256), 256, 0, stream>>>(s, d, n);
  };

  // ---- one-time (per call) weight conversion to bf16 ----
  cvt(win_w,   winBf,   12L * 3 * ED * HD);
  cvt(wbeta_w, wbetaBf, 12L * ED * HD);
  cvt(wout_w,  woutBf,  12L * HD * ED);
  cvt(mlp1_w,  m1Bf,    12L * 4 * HD * HD);
  cvt(mlp2_w,  m2Bf,    12L * 4 * HD * HD);
  cvt(pe2_w,   pe2Bf,   384L * 1728);
  cvt(head_w,  headBf,  1000L * 384);

  // ---- patch embedding ----
  {
    int tot = BD * 192 * 64 * 64;
    conv1_kernel<<<(tot + 255) / 256, 256, 0, stream>>>(x, pe1_w, pe1_b, h1);
    tot = BD * 1024 * 1728;
    im2col_kernel<<<(tot + 255) / 256, 256, 0, stream>>>(h1, colBf);
    // conv2 as GEMM: per-batch (1024 x 1728) @ (384 x 1728)^T -> h rows 1..1024
    launch_gemm(stream, colBf, 1728, 1, 1024L * 1728,
                pe2Bf, 1728, 1, 0,
                pe2_b, nullptr, 0, 0,
                h + HD, HD, (long)LD * HD, 1024, HD, 1728, BD, 0, 0);
    tot = (int)(BD * LD * HD);
    embed_kernel<<<(tot + 255) / 256, 256, 0, stream>>>(h, cls_tok, pos_emb);
  }

  // ---- transformer blocks ----
  for (int i = 0; i < 12; ++i) {
    const float* g1 = ln1_g + (long)i * HD;  const float* b1 = ln1_b + (long)i * HD;
    const float* g2 = ln2_g + (long)i * ED;  const float* b2 = ln2_b + (long)i * ED;

    // LN1 -> xn (bf16)
    ln_kernel<<<(int)((BL + 7) / 8), 256, 0, stream>>>(h, HD, xnBf, HD, g1, b1,
                                                       nullptr, (int)BL, HD);

    // qkv = xn @ win^T + b    (8200 x 2304, K=384) -> f32
    launch_gemm(stream, xnBf, HD, 1, 0,
                winBf + (long)i * 3 * ED * HD, HD, 1, 0,
                win_b + (long)i * 3 * ED, nullptr, 0, 0,
                qkvA, E3, 0, (int)BL, 3 * ED, HD, 1, 0, 0);

    // beta = sigmoid(xn @ wbeta^T + b)*0.9+0.1 (act=2) -> f32
    launch_gemm(stream, xnBf, HD, 1, 0,
                wbetaBf + (long)i * ED * HD, HD, 1, 0,
                wbeta_b + (long)i * ED, nullptr, 0, 0,
                betaB, ED, 0, (int)BL, ED, HD, 1, 2, 0);

    // silu/silu/gelu + coupled l2norm (in-place f32)
    qkvact_kernel<<<(int)((BL + 7) / 8), 256, 0, stream>>>(qkvA, (int)BL);

    // depthwise token conv + beta-gated v -> qkvBf (bf16)
    {
      int tot = (int)(BL * E3);
      dwconv_kernel<<<(tot + 255) / 256, 256, 0, stream>>>(
          qkvA, qkvBf, conv_w + (long)i * ED * 3, betaB, tot);
    }

    // state[b] = v_new^T @ k   (E x E, K=L, batched, m-contiguous) -> bf16
    launch_gemm(stream, qkvBf + 2 * ED, 1, E3, (long)LD * E3,
                qkvBf + ED, 1, E3, (long)LD * E3,
                nullptr, nullptr, 0, 0,
                (float*)stateBf, ED, (long)ED * ED, ED, ED, LD, BD, 0, 1);

    // out[b] = q @ state[b]^T  (L x E, K=E, batched) -> f32
    launch_gemm(stream, qkvBf, E3, 1, (long)LD * E3,
                stateBf, ED, 1, (long)ED * ED,
                nullptr, nullptr, 0, 0,
                attnO, ED, (long)LD * ED, LD, ED, ED, BD, 0, 0);

    // scale + LN2 -> attnBf (bf16)
    ln_kernel<<<(int)((BL + 7) / 8), 256, 0, stream>>>(attnO, ED, attnBf, ED,
                                                       g2, b2, attn_sc + i,
                                                       (int)BL, ED);

    // h += attnBf @ wout^T + b   (residual fused)
    launch_gemm(stream, attnBf, ED, 1, 0,
                woutBf + (long)i * HD * ED, ED, 1, 0,
                wout_b + (long)i * HD, h, HD, 0,
                h, HD, 0, (int)BL, HD, ED, 1, 0, 0);

    // MLP branch: LN1 params again
    ln_kernel<<<(int)((BL + 7) / 8), 256, 0, stream>>>(h, HD, xnBf, HD, g1, b1,
                                                       nullptr, (int)BL, HD);

    // mid = gelu(xn @ m1^T + b)  (act=1) -> bf16
    launch_gemm(stream, xnBf, HD, 1, 0,
                m1Bf + (long)i * 4 * HD * HD, HD, 1, 0,
                mlp1_b + (long)i * 4 * HD, nullptr, 0, 0,
                (float*)midBf, 4 * HD, 0, (int)BL, 4 * HD, HD, 1, 1, 1);

    // h += mid @ m2^T + b  (residual fused)
    launch_gemm(stream, midBf, 4 * HD, 1, 0,
                m2Bf + (long)i * 4 * HD * HD, 4 * HD, 1, 0,
                mlp2_b + (long)i * HD, h, HD, 0,
                h, HD, 0, (int)BL, HD, 4 * HD, 1, 0, 0);
  }

  // ---- final LN on cls rows + head ----
  ln_kernel<<<1, 256, 0, stream>>>(h, (long)LD * HD, clsnBf, HD, lnf_g, lnf_b,
                                   nullptr, BD, HD);
  launch_gemm(stream, clsnBf, HD, 1, 0,
              headBf, HD, 1, 0,
              head_b, nullptr, 0, 0,
              out, 1000, 0, BD, 1000, HD, 1, 0, 0);

  (void)in_sizes; (void)n_in; (void)out_size; (void)ws_size;
}
